// efficient_Attention_modified_Param_sharing_21071109554260
// MI455X (gfx1250) — compile-verified, matrix operand order checked by reading, not run on hardware
//
#include <hip/hip_runtime.h>
#include <hip/hip_bf16.h>

// ---------------------------------------------------------------------------
// efficient attention (param sharing) for MI455X / gfx1250, wave32 + WMMA bf16
// B=8, DIM=256, N=H*W=16384, HEADS=8, D=32
// ---------------------------------------------------------------------------

typedef __bf16 bf;
typedef __attribute__((ext_vector_type(16))) __bf16 v16bf;
typedef __attribute__((ext_vector_type(8)))  float  v8f;

#define WMMA_BF16(A, B, C) \
    __builtin_amdgcn_wmma_f32_16x16x32_bf16(false, (A), false, (B), (short)0, (C), false, false)

static constexpr int NB    = 8;       // batch
static constexpr int DIM   = 256;
static constexpr int NSP   = 16384;   // H*W
static constexpr int HEADS = 8;
static constexpr int DD    = 32;      // qkv dim

// workspace layout (bytes)
static constexpr size_t WS_QSM   = 0;                                   // bf16 [B][256][N]
static constexpr size_t WS_KVW   = WS_QSM  + (size_t)NB*DIM*NSP*2;      // f32  [B][32][N]
static constexpr size_t WS_KEYSM = WS_KVW  + (size_t)NB*DD*NSP*4;       // bf16 [B][32][N]
static constexpr size_t WS_KVBF  = WS_KEYSM+ (size_t)NB*DD*NSP*2;       // bf16 [B][32][N]
static constexpr size_t WS_CTX   = WS_KVBF + (size_t)NB*DD*NSP*2;       // f32  [B][32][32]

// Load a 16(row)x32(K) bf16 WMMA operand tile from row-major storage.
// Lane l -> row (l&15); K offset ((l>>4)<<3); VGPR v holds K = off+16*(v>>2)+2*(v&3) (+1).
// Used for A operands directly, and for B operands from transposed (T[n][k]) storage.
static __device__ __forceinline__ v16bf load_op16x32(const bf* base, int ld, int lane) {
    const bf* rp = base + (size_t)(lane & 15) * (size_t)ld + ((lane >> 4) << 3);
    v16bf r;
#pragma unroll
    for (int v = 0; v < 8; ++v) {
        int k = ((v >> 2) << 4) + ((v & 3) << 1);
        r[2 * v]     = rp[k];
        r[2 * v + 1] = rp[k + 1];
    }
    return r;
}

static __device__ __forceinline__ v8f vzero8() {
    v8f z = {0.f, 0.f, 0.f, 0.f, 0.f, 0.f, 0.f, 0.f};
    return z;
}

// ---------------------------------------------------------------------------
// K1: [q;kv] = W_aug(288x256) @ x_tile(256x128), BN affine, softmax over D for q.
// Grid: (N/128, B), block 256 (8 waves). Wave w owns local cols 16w..16w+15.
// ---------------------------------------------------------------------------
__global__ void __launch_bounds__(256)
k1_proj_qsoftmax(const float* __restrict__ x,  const float* __restrict__ wq,
                 const float* __restrict__ wkv,
                 const float* __restrict__ qs, const float* __restrict__ qb,
                 const float* __restrict__ kvs,const float* __restrict__ kvb,
                 bf* __restrict__ qsm, float* __restrict__ kvw) {
    __shared__ bf Wl[288][32];   // 18 KB, row-major (A operand)
    __shared__ bf Xl[128][32];   // 8 KB, transposed x: Xl[n][k] (B operand)

    const int tid  = threadIdx.x;
    const int lane = tid & 31;
    const int wv   = tid >> 5;
    const int b    = blockIdx.y;
    const int n0   = blockIdx.x * 128;

    v8f acc[18];
#pragma unroll
    for (int t = 0; t < 18; ++t) acc[t] = vzero8();

    for (int ks = 0; ks < 8; ++ks) {
        const int k0 = ks * 32;
        // stage W_aug slice [288][32] -> bf16 LDS
        for (int i = tid; i < 288 * 32; i += 256) {
            int r = i >> 5, kk = i & 31;
            float w = (r < 256) ? wq[r * 256 + k0 + kk] : wkv[(r - 256) * 256 + k0 + kk];
            Wl[r][kk] = (bf)w;
        }
        // stage x slice transposed: Xl[n][k]
        const int nloc = tid & 127;
#pragma unroll 4
        for (int i = 0; i < 16; ++i) {
            int kk = (tid >> 7) + 2 * i;
            Xl[nloc][kk] = (bf)x[(size_t)(b * 256 + k0 + kk) * NSP + n0 + nloc];
        }
        if (ks < 7)  // CDNA5 prefetch of next K slice (global_prefetch_b8)
            __builtin_prefetch(&x[(size_t)(b * 256 + k0 + 32 + (tid >> 7)) * NSP + n0 + nloc], 0, 0);
        __syncthreads();

        v16bf bop = load_op16x32(&Xl[16 * wv][0], 32, lane);
#pragma unroll
        for (int t = 0; t < 18; ++t) {
            v16bf aop = load_op16x32(&Wl[16 * t][0], 32, lane);
            acc[t] = WMMA_BF16(aop, bop, acc[t]);
        }
        __syncthreads();
    }

    // Epilogue. C layout: lane l holds col (l&15), rows v + 8*(l>>4) of each 16x16 tile.
    const int hi   = lane >> 4;
    const int gcol = n0 + 16 * wv + (lane & 15);

    // q rows (tiles 0..15): affine, then softmax over the 32 channels of each head.
#pragma unroll
    for (int h = 0; h < HEADS; ++h) {
        float v0[8], v1[8];
        float m = -3.0e38f;
#pragma unroll
        for (int v = 0; v < 8; ++v) {
            int r0 = 32 * h + 8 * hi + v;
            int r1 = 32 * h + 16 + 8 * hi + v;
            v0[v] = acc[2 * h][v]     * qs[r0] + qb[r0];
            v1[v] = acc[2 * h + 1][v] * qs[r1] + qb[r1];
            m = fmaxf(m, fmaxf(v0[v], v1[v]));
        }
        m = fmaxf(m, __shfl_xor(m, 16, 32));  // lanes l and l^16 share a column
        float s = 0.f;
#pragma unroll
        for (int v = 0; v < 8; ++v) {
            v0[v] = __expf(v0[v] - m);
            v1[v] = __expf(v1[v] - m);
            s += v0[v] + v1[v];
        }
        s += __shfl_xor(s, 16, 32);
        float inv = __frcp_rn(s);
#pragma unroll
        for (int v = 0; v < 8; ++v) {
            int r0 = 32 * h + 8 * hi + v;
            int r1 = 32 * h + 16 + 8 * hi + v;
            qsm[(size_t)(b * 256 + r0) * NSP + gcol] = (bf)(v0[v] * inv);
            qsm[(size_t)(b * 256 + r1) * NSP + gcol] = (bf)(v1[v] * inv);
        }
    }
    // kv rows (tiles 16,17): affine, store fp32
#pragma unroll
    for (int t = 16; t < 18; ++t) {
#pragma unroll
        for (int v = 0; v < 8; ++v) {
            int r = 16 * (t - 16) + 8 * hi + v;
            float val = acc[t][v] * kvs[r] + kvb[r];
            kvw[(size_t)(b * 32 + r) * NSP + gcol] = val;
        }
    }
}

// ---------------------------------------------------------------------------
// K2a: per-(b,d) row softmax stats over N=16384; emit key_sm (bf16) + kv (bf16).
// Grid (32, B), 256 threads.
// ---------------------------------------------------------------------------
__global__ void __launch_bounds__(256)
k2a_rowsoftmax(const float* __restrict__ kvw, bf* __restrict__ keysm, bf* __restrict__ kvbf) {
    __shared__ float red[256];
    const int tid = threadIdx.x;
    const size_t base = (size_t)(blockIdx.y * 32 + blockIdx.x) * NSP;

    float m = -3.0e38f;
    for (int i = tid; i < NSP; i += 256) m = fmaxf(m, kvw[base + i]);
    red[tid] = m; __syncthreads();
    for (int s = 128; s > 0; s >>= 1) {
        if (tid < s) red[tid] = fmaxf(red[tid], red[tid + s]);
        __syncthreads();
    }
    const float mx = red[0];
    __syncthreads();

    float s = 0.f;
    for (int i = tid; i < NSP; i += 256) s += __expf(kvw[base + i] - mx);
    red[tid] = s; __syncthreads();
    for (int t = 128; t > 0; t >>= 1) {
        if (tid < t) red[tid] += red[tid + t];
        __syncthreads();
    }
    const float inv = __frcp_rn(red[0]);

    for (int i = tid; i < NSP; i += 256) {
        float v = kvw[base + i];
        keysm[base + i] = (bf)(__expf(v - mx) * inv);
        kvbf[base + i]  = (bf)v;
    }
}

// ---------------------------------------------------------------------------
// K2b: context[b] = key_sm(32xN) @ kv^T(Nx32).  Grid (B), 256 threads;
// wave w reduces K slice [2048w, 2048w+2048), LDS ds_add_f32 cross-wave reduce.
// ---------------------------------------------------------------------------
__global__ void __launch_bounds__(256)
k2b_context(const bf* __restrict__ keysm, const bf* __restrict__ kvbf, float* __restrict__ ctx) {
    __shared__ float cl[32 * 32];
    const int tid = threadIdx.x, lane = tid & 31, wv = tid >> 5;
    const int b = blockIdx.x;

    for (int i = tid; i < 1024; i += 256) cl[i] = 0.f;
    __syncthreads();

    v8f a00 = vzero8(), a01 = vzero8(), a10 = vzero8(), a11 = vzero8();
    const bf* arow0 = keysm + (size_t)(b * 32) * NSP;
    const bf* brow0 = kvbf  + (size_t)(b * 32) * NSP;
    for (int ks = 0; ks < 64; ++ks) {
        const int k = wv * 2048 + 32 * ks;
        v16bf A0 = load_op16x32(arow0 + k,                    NSP, lane);
        v16bf A1 = load_op16x32(arow0 + (size_t)16 * NSP + k, NSP, lane);
        v16bf B0 = load_op16x32(brow0 + k,                    NSP, lane);
        v16bf B1 = load_op16x32(brow0 + (size_t)16 * NSP + k, NSP, lane);
        a00 = WMMA_BF16(A0, B0, a00);
        a01 = WMMA_BF16(A0, B1, a01);
        a10 = WMMA_BF16(A1, B0, a10);
        a11 = WMMA_BF16(A1, B1, a11);
    }
    const int hi = lane >> 4, e0 = lane & 15;
#pragma unroll
    for (int v = 0; v < 8; ++v) {
        int d = 8 * hi + v;
        atomicAdd(&cl[(d)      * 32 + e0],      a00[v]);
        atomicAdd(&cl[(d)      * 32 + e0 + 16], a01[v]);
        atomicAdd(&cl[(d + 16) * 32 + e0],      a10[v]);
        atomicAdd(&cl[(d + 16) * 32 + e0 + 16], a11[v]);
    }
    __syncthreads();
    for (int i = tid; i < 1024; i += 256) ctx[b * 1024 + i] = cl[i];
}

// ---------------------------------------------------------------------------
// K3: attended = ctx^T @ qsm (per head, K=32, one WMMA/tile) -> ReLU ->
//     out = wp(256x256) @ attended, + p affine.  Grid (N/64, B), 256 threads.
// Wave w == head w in phase A; owns row tiles {2w,2w+1} in phase B.
// ---------------------------------------------------------------------------
__global__ void __launch_bounds__(256)
k3_attend_out(const bf* __restrict__ qsm, const float* __restrict__ ctx,
              const float* __restrict__ wp, const float* __restrict__ ps,
              const float* __restrict__ pb, float* __restrict__ out) {
    __shared__ bf Qsm_t[64][256];  // 32 KB, transposed: [col][row]
    __shared__ bf Att_t[64][256];  // 32 KB, transposed: [col][row]

    const int tid = threadIdx.x, lane = tid & 31, wv = tid >> 5;
    const int b = blockIdx.y;
    const int n0 = blockIdx.x * 64;
    const int hi = lane >> 4, lc = lane & 15;

    // stage qsm tile transposed into LDS
    for (int i = tid; i < 256 * 64; i += 256) {
        int r = i >> 6, c = i & 63;
        Qsm_t[c][r] = qsm[(size_t)(b * 256 + r) * NSP + n0 + c];
    }
    __syncthreads();

    // ---- phase A: attended + ReLU (head h = wv) ----
    const float* cb = ctx + b * 1024;
    v16bf aop[2];
#pragma unroll
    for (int j = 0; j < 2; ++j) {
        int e = 16 * j + lc;  // A[m][k] = ctx[k][e]
#pragma unroll
        for (int v = 0; v < 8; ++v) {
            int k = ((lane >> 4) << 3) + ((v >> 2) << 4) + ((v & 3) << 1);
            aop[j][2 * v]     = (bf)cb[k * 32 + e];
            aop[j][2 * v + 1] = (bf)cb[(k + 1) * 32 + e];
        }
    }
#pragma unroll
    for (int ct = 0; ct < 4; ++ct) {
        v16bf bop = load_op16x32(&Qsm_t[16 * ct][32 * wv], 256, lane);
#pragma unroll
        for (int j = 0; j < 2; ++j) {
            v8f c = vzero8();
            c = WMMA_BF16(aop[j], bop, c);   // K = 32 in one shot
            int rt = 2 * wv + j;
#pragma unroll
            for (int v = 0; v < 8; ++v) {
                int row = 16 * rt + 8 * hi + v;
                int col = 16 * ct + lc;
                Att_t[col][row] = (bf)fmaxf(c[v], 0.f);
            }
        }
    }
    __syncthreads();

    // ---- phase B: out = wp @ attended ----
    v8f acc[2][4];
#pragma unroll
    for (int i = 0; i < 2; ++i)
#pragma unroll
        for (int ct = 0; ct < 4; ++ct) acc[i][ct] = vzero8();

    for (int ks = 0; ks < 8; ++ks) {
        v16bf aw[2];
#pragma unroll
        for (int i = 0; i < 2; ++i) {
            int row = 16 * (2 * wv + i) + lc;
            const float* wr = wp + row * 256 + 32 * ks + ((lane >> 4) << 3);
#pragma unroll
            for (int v = 0; v < 8; ++v) {
                int kk = ((v >> 2) << 4) + ((v & 3) << 1);
                aw[i][2 * v]     = (bf)wr[kk];
                aw[i][2 * v + 1] = (bf)wr[kk + 1];
            }
        }
#pragma unroll
        for (int ct = 0; ct < 4; ++ct) {
            v16bf bop = load_op16x32(&Att_t[16 * ct][32 * ks], 256, lane);
            acc[0][ct] = WMMA_BF16(aw[0], bop, acc[0][ct]);
            acc[1][ct] = WMMA_BF16(aw[1], bop, acc[1][ct]);
        }
    }
#pragma unroll
    for (int i = 0; i < 2; ++i)
#pragma unroll
        for (int ct = 0; ct < 4; ++ct)
#pragma unroll
            for (int v = 0; v < 8; ++v) {
                int row = 16 * (2 * wv + i) + 8 * hi + v;
                int gc  = n0 + 16 * ct + lc;
                out[(size_t)(b * 256 + row) * NSP + gc] = acc[i][ct][v] * ps[row] + pb[row];
            }
}

// ---------------------------------------------------------------------------
extern "C" void kernel_launch(void* const* d_in, const int* in_sizes, int n_in,
                              void* d_out, int out_size, void* d_ws, size_t ws_size,
                              hipStream_t stream) {
    const float* x   = (const float*)d_in[0];
    const float* wq  = (const float*)d_in[1];
    const float* qs  = (const float*)d_in[2];
    const float* qb  = (const float*)d_in[3];
    const float* wkv = (const float*)d_in[4];
    const float* kvs = (const float*)d_in[5];
    const float* kvb = (const float*)d_in[6];
    const float* wp  = (const float*)d_in[7];
    const float* ps  = (const float*)d_in[8];
    const float* pb  = (const float*)d_in[9];
    float* out = (float*)d_out;

    char* ws = (char*)d_ws;
    bf*    qsm   = (bf*)(ws + WS_QSM);
    float* kvw   = (float*)(ws + WS_KVW);
    bf*    keysm = (bf*)(ws + WS_KEYSM);
    bf*    kvbf  = (bf*)(ws + WS_KVBF);
    float* ctx   = (float*)(ws + WS_CTX);

    k1_proj_qsoftmax<<<dim3(NSP / 128, NB), 256, 0, stream>>>(x, wq, wkv, qs, qb, kvs, kvb, qsm, kvw);
    k2a_rowsoftmax  <<<dim3(DD, NB),        256, 0, stream>>>(kvw, keysm, kvbf);
    k2b_context     <<<dim3(NB),            256, 0, stream>>>(keysm, kvbf, ctx);
    k3_attend_out   <<<dim3(NSP / 64, NB),  256, 0, stream>>>(qsm, ctx, wp, ps, pb, out);
}
